// TemporalAttention_66065186947625
// MI455X (gfx1250) — matrix operand order repaired and checked by reading, not compiled, not run on hardware
//
#include <hip/hip_runtime.h>

// TemporalAttention for MI455X (gfx1250): bf16 WMMA flash attention, f32 accumulate.
// ~150 GFLOP vs ~34 MB input -> compute bound -> v_wmma_f32_16x16x32_bf16 everywhere.
// K/V tiles staged per-workgroup into LDS by the Tensor Data Mover
// (tensor_load_to_lds, TENSORcnt), double-buffered; fragments fed via ds_load_b128.

#define BB 8
#define NN 4096
#define DD 256

typedef __attribute__((ext_vector_type(16))) __bf16 v16bf;
typedef __attribute__((ext_vector_type(8)))  float  v8f;
typedef __attribute__((ext_vector_type(8)))  unsigned short ushort8;
typedef __attribute__((ext_vector_type(4)))  unsigned int  uint4v;
typedef __attribute__((ext_vector_type(8)))  unsigned int  uint8v;

union Frag {
  v16bf   bf;
  ushort8 u[2];
};

__device__ __forceinline__ unsigned short f2bf(float f) {
  unsigned u = __builtin_bit_cast(unsigned, f);
  u += 0x7FFFu + ((u >> 16) & 1u);            // round-to-nearest-even
  return (unsigned short)(u >> 16);
}

__device__ __forceinline__ v8f wmma_bf16(const Frag& a, const Frag& b, v8f c) {
  return __builtin_amdgcn_wmma_f32_16x16x32_bf16(false, a.bf, false, b.bf,
                                                 (short)0, c, false, false);
}

// TDM: 2-D tile load, descriptor groups 0 (4 SGPRs) + 1 (8 SGPRs), groups 2/3 NULL.
__device__ __forceinline__ void tdm_load_2d(unsigned ldsByteAddr,
                                            unsigned long long globalAddr,
                                            unsigned dim0,      // elements per row
                                            unsigned dim1,      // rows
                                            unsigned stride0) { // row stride (elems)
  uint4v g0;
  g0.x = 1u;                                        // count=1 user descriptor
  g0.y = ldsByteAddr;                               // lds_addr [63:32]
  g0.z = (unsigned)(globalAddr & 0xffffffffu);      // global_addr [95:64]
  g0.w = (unsigned)(globalAddr >> 32) | 0x80000000u;// addr[56:32] | type=2
  uint8v g1;
  g1.s0 = 0x00010000u;            // workgroup_mask=0, data_size=1 (2 bytes)
  g1.s1 = dim0 << 16;             // tensor_dim0[15:0]  (bits 79:48)
  g1.s2 = (dim0 >> 16) | (dim1 << 16);   // dim0 hi | tensor_dim1 lo
  g1.s3 = (dim1 >> 16) | (dim0 << 16);   // dim1 hi | tile_dim0 (= dim0)
  g1.s4 = dim1;                   // tile_dim1 (= dim1), tile_dim2 = 0
  g1.s5 = stride0;                // tensor_dim0_stride [207:160] lo
  g1.s6 = 0u;                     // stride0 hi | tensor_dim1_stride lo
  g1.s7 = 0u;
  asm volatile("tensor_load_to_lds %0, %1" :: "s"(g0), "s"(g1) : "memory");
}

// ---------------- conversion kernels ----------------

__global__ void TA_cvt_f32_bf16(const float* __restrict__ src,
                                unsigned short* __restrict__ dst, long n) {
  long i = (long)blockIdx.x * blockDim.x + threadIdx.x;
  if (i < n) dst[i] = f2bf(src[i]);
}

// WT[e, d] = bf16(W[d, e])  (D x D)
__global__ void TA_cvt_transpose_bf16(const float* __restrict__ W,
                                      unsigned short* __restrict__ WT) {
  int i = blockIdx.x * blockDim.x + threadIdx.x;   // over D*D
  int d = i >> 8;
  int e = i & (DD - 1);
  WT[(size_t)e * DD + d] = f2bf(W[i]);
}

// ---------------- projection GEMM: Q/K/V = x @ W + b ----------------
// block = 128 threads = 4 waves; wave computes one 16x16 tile.
// z = 0:Q  1:K  2:V (V stored transposed [B, D, N])

__global__ void TA_proj_wmma(const unsigned short* __restrict__ xb,
                             const unsigned short* __restrict__ WTall,
                             const float* __restrict__ bq,
                             const float* __restrict__ bk,
                             const float* __restrict__ bv,
                             unsigned short* __restrict__ Qb,
                             unsigned short* __restrict__ Kb,
                             unsigned short* __restrict__ Vt) {
  const int lane = threadIdx.x & 31;
  const int wid  = threadIdx.x >> 5;
  const int l16  = lane & 15;
  const int half = lane >> 4;
  const int z    = blockIdx.z;
  const int m0   = blockIdx.x * 16;                 // row tile in [0, B*N)
  const int n0   = (blockIdx.y * 4 + wid) * 16;     // output-column tile

  const unsigned short* WT   = WTall + (size_t)z * DD * DD;
  const float*          bias = (z == 0) ? bq : (z == 1) ? bk : bv;

  const unsigned short* arow = xb + (size_t)(m0 + l16) * DD;             // A row
  const unsigned short* brow = WT + (size_t)(n0 + l16) * DD + half * 16; // B col

  v8f c = {};
#pragma unroll
  for (int d = 0; d < 8; ++d) {
    Frag a, b;
    a.u[0] = *reinterpret_cast<const ushort8*>(arow + d * 32 + half * 16);
    a.u[1] = *reinterpret_cast<const ushort8*>(arow + d * 32 + half * 16 + 8);
    b.u[0] = *reinterpret_cast<const ushort8*>(brow + d * 32);
    b.u[1] = *reinterpret_cast<const ushort8*>(brow + d * 32 + 8);
    c = wmma_bf16(a, b, c);
  }

  const float bvv = bias[n0 + l16];
#pragma unroll
  for (int j = 0; j < 8; ++j) {
    const int r = m0 + j + 8 * half;                // global row in B*N
    const unsigned short val = f2bf(c[j] + bvv);
    if (z == 0) {
      Qb[(size_t)r * DD + n0 + l16] = val;
    } else if (z == 1) {
      Kb[(size_t)r * DD + n0 + l16] = val;
    } else {
      const int bI = r >> 12;                        // r / N
      const int nI = r & (NN - 1);                   // r % N
      Vt[((size_t)bI * DD + n0 + l16) * NN + nI] = val;
    }
  }
}

// ---------------- fused flash attention ----------------
// block = 256 threads = 8 waves; wave owns 16 q-rows, full D=256 accumulator.
// Dynamic LDS layout (bytes):
//   [0,     32768)  Ks double buffer: 2 x (32 keys x 256 d)  bf16
//   [32768, 65536)  Vs double buffer: 2 x (256 d x 32 k)     bf16 (V transposed)
//   [65536, 73728)  per-wave P staging: 8 x (16 x 32)        bf16
// total 73728 bytes.

__global__ void __launch_bounds__(256, 1)
TA_flash_attn_wmma(const unsigned short* __restrict__ Qb,
                   const unsigned short* __restrict__ Kb,
                   const unsigned short* __restrict__ Vt,
                   float* __restrict__ out) {
  extern __shared__ unsigned short smem[];

  const int tid  = threadIdx.x;
  const int lane = tid & 31;
  const int wid  = tid >> 5;
  const int l16  = lane & 15;
  const int half = lane >> 4;
  const int b    = blockIdx.y;
  const int q0   = blockIdx.x * 128 + wid * 16;

  unsigned short* pb = smem + 32768 + wid * 512;   // ushort index 32768 = byte 65536

  const unsigned long long kbase =
      (unsigned long long)(Kb + (size_t)b * NN * DD);
  const unsigned long long vbase =
      (unsigned long long)(Vt + (size_t)b * DD * NN);

  // stage one 32-key step via TDM: K tile 32x256 + V^T tile 256x32 (16KB each)
  auto tdm_stage = [&](int buf, int k0) {
    // K: rows = keys (32), row length 256 elems, row stride 256 elems
    tdm_load_2d((unsigned)(buf * 16384),
                kbase + (unsigned long long)k0 * (DD * 2),
                DD, 32u, DD);
    // V^T: rows = d (256), row length 32 elems, row stride NN elems
    tdm_load_2d((unsigned)(32768 + buf * 16384),
                vbase + (unsigned long long)k0 * 2,
                32u, DD, NN);
  };

  // fold 1/sqrt(D) and log2(e): softmax computed with exp2
  const float cs = 0.0625f * 1.4426950408889634f;

  // preload Q fragments: 16 rows x 256 d -> 8 A-fragments (kept in VGPRs)
  Frag qf[8];
  const unsigned short* qrow = Qb + ((size_t)b * NN + q0 + l16) * DD;
#pragma unroll
  for (int d = 0; d < 8; ++d) {
    qf[d].u[0] = *reinterpret_cast<const ushort8*>(qrow + d * 32 + half * 16);
    qf[d].u[1] = *reinterpret_cast<const ushort8*>(qrow + d * 32 + half * 16 + 8);
  }

  float m2[8], lsum[8];
  v8f   acc[16];
  const v8f vzero = {};
#pragma unroll
  for (int j = 0; j < 8; ++j) { m2[j] = -3.0e38f; lsum[j] = 0.0f; }
#pragma unroll
  for (int t = 0; t < 16; ++t) acc[t] = vzero;

  if (wid == 0) tdm_stage(0, 0);

  for (int it = 0; it < NN / 32; ++it) {
    const int buf = it & 1;
    if (wid == 0) __builtin_amdgcn_s_wait_tensorcnt(0);  // DMA landed (wave 0)
    __syncthreads();                                     // tile visible block-wide
    if (wid == 0 && it + 1 < NN / 32) tdm_stage(buf ^ 1, (it + 1) * 32);

    const unsigned short* KsBuf = smem + buf * 8192;
    const unsigned short* VsBuf = smem + 16384 + buf * 8192;

    // ---- S = Q K^T for 32 keys (two 16-wide C tiles) ----
    v8f s0 = {}, s1 = {};
#pragma unroll
    for (int d = 0; d < 8; ++d) {
      const unsigned short* kp0 = KsBuf + l16 * 256 + d * 32 + half * 16;
      const unsigned short* kp1 = KsBuf + (16 + l16) * 256 + d * 32 + half * 16;
      Frag kf0, kf1;
      kf0.u[0] = *reinterpret_cast<const ushort8*>(kp0);
      kf0.u[1] = *reinterpret_cast<const ushort8*>(kp0 + 8);
      kf1.u[0] = *reinterpret_cast<const ushort8*>(kp1);
      kf1.u[1] = *reinterpret_cast<const ushort8*>(kp1 + 8);
      s0 = wmma_bf16(qf[d], kf0, s0);
      s1 = wmma_bf16(qf[d], kf1, s1);
    }

    // ---- online softmax (rows per-VGPR j; cols across 16-lane group) ----
    float alpha[8];
#pragma unroll
    for (int j = 0; j < 8; ++j) {
      const float a0 = s0[j] * cs;
      const float a1 = s1[j] * cs;
      float v = fmaxf(a0, a1);
      v = fmaxf(v, __shfl_xor(v, 1));
      v = fmaxf(v, __shfl_xor(v, 2));
      v = fmaxf(v, __shfl_xor(v, 4));
      v = fmaxf(v, __shfl_xor(v, 8));
      const float mnew = fmaxf(m2[j], v);
      const float p0 = __builtin_amdgcn_exp2f(a0 - mnew);
      const float p1 = __builtin_amdgcn_exp2f(a1 - mnew);
      alpha[j] = __builtin_amdgcn_exp2f(m2[j] - mnew);
      m2[j] = mnew;
      float rs = p0 + p1;
      rs += __shfl_xor(rs, 1);
      rs += __shfl_xor(rs, 2);
      rs += __shfl_xor(rs, 4);
      rs += __shfl_xor(rs, 8);
      lsum[j] = lsum[j] * alpha[j] + rs;
      // stage P (C layout -> LDS) for A-layout reload
      pb[(j + 8 * half) * 32 + l16]      = f2bf(p0);
      pb[(j + 8 * half) * 32 + 16 + l16] = f2bf(p1);
    }

    // rescale accumulators
#pragma unroll
    for (int t = 0; t < 16; ++t) {
#pragma unroll
      for (int j = 0; j < 8; ++j) acc[t][j] *= alpha[j];
    }

    // ---- reload P as 16x32 A fragment (same-wave LDS, in-order DS) ----
    Frag pf;
    pf.u[0] = *reinterpret_cast<const ushort8*>(pb + l16 * 32 + half * 16);
    pf.u[1] = *reinterpret_cast<const ushort8*>(pb + l16 * 32 + half * 16 + 8);

    // ---- O += P * V (V^T tile in LDS: contiguous k per lane) ----
#pragma unroll
    for (int t = 0; t < 16; ++t) {
      const unsigned short* vp = VsBuf + (t * 16 + l16) * 32 + half * 16;
      Frag vf;
      vf.u[0] = *reinterpret_cast<const ushort8*>(vp);
      vf.u[1] = *reinterpret_cast<const ushort8*>(vp + 8);
      acc[t] = wmma_bf16(pf, vf, acc[t]);
    }

    __syncthreads();      // all waves done reading this buffer
  }

  // ---- epilogue: O / l ----
#pragma unroll
  for (int j = 0; j < 8; ++j) {
    const float inv = 1.0f / lsum[j];
    float* orow = out + ((size_t)b * NN + q0 + j + 8 * half) * DD + l16;
#pragma unroll
    for (int t = 0; t < 16; ++t) orow[t * 16] = acc[t][j] * inv;
  }
}

// ---------------- launch ----------------

extern "C" void kernel_launch(void* const* d_in, const int* in_sizes, int n_in,
                              void* d_out, int out_size, void* d_ws, size_t ws_size,
                              hipStream_t stream) {
  const float* x  = (const float*)d_in[0];
  const float* Wq = (const float*)d_in[1];
  const float* bq = (const float*)d_in[2];
  const float* Wk = (const float*)d_in[3];
  const float* bk = (const float*)d_in[4];
  const float* Wv = (const float*)d_in[5];
  const float* bv = (const float*)d_in[6];
  float* out = (float*)d_out;

  const long szX = (long)BB * NN * DD;   // 8,388,608 elements
  // workspace layout (bf16 = ushort): xb | WqT,WkT,WvT | Qb | Kb | Vt  (~67.5 MB)
  unsigned short* ws = (unsigned short*)d_ws;
  unsigned short* xb = ws;
  unsigned short* WT = xb + szX;
  unsigned short* Qb = WT + 3L * DD * DD;
  unsigned short* Kb = Qb + szX;
  unsigned short* Vt = Kb + szX;

  TA_cvt_f32_bf16<<<(int)((szX + 255) / 256), 256, 0, stream>>>(x, xb, szX);
  TA_cvt_transpose_bf16<<<(DD * DD) / 256, 256, 0, stream>>>(Wq, WT);
  TA_cvt_transpose_bf16<<<(DD * DD) / 256, 256, 0, stream>>>(Wk, WT + DD * DD);
  TA_cvt_transpose_bf16<<<(DD * DD) / 256, 256, 0, stream>>>(Wv, WT + 2 * DD * DD);

  dim3 pg(BB * NN / 16, DD / 64, 3);     // 2048 x 4 x 3, 4 waves/block
  TA_proj_wmma<<<pg, 128, 0, stream>>>(xb, WT, bq, bk, bv, Qb, Kb, Vt);

  dim3 ag(NN / 128, BB);                 // 32 x 8 blocks, 8 waves/block
  const size_t smemBytes = 2 * 16384 + 2 * 16384 + 8 * 1024;  // 73728 bytes
  TA_flash_attn_wmma<<<ag, 256, smemBytes, stream>>>(Qb, Kb, Vt, out);
}
